// GlobalLocalAttention_9543417332420
// MI455X (gfx1250) — matrix operand order repaired and checked by reading, not compiled, not run on hardware
//
#include <hip/hip_runtime.h>
#include <hip/hip_bf16.h>

typedef __attribute__((ext_vector_type(16))) __bf16 v16bf;
typedef __attribute__((ext_vector_type(8)))  __bf16 v8bf;
typedef __attribute__((ext_vector_type(8)))  float  v8f;
typedef int v4i_vs __attribute__((vector_size(16)));
typedef __attribute__((address_space(1))) v4i_vs* as1_v4i;
typedef __attribute__((address_space(3))) v4i_vs* as3_v4i;

static constexpr int Bsz = 2;
static constexpr int C   = 256;
static constexpr int H   = 96;
static constexpr int W   = 96;
static constexpr int HW  = H * W;          // 9216
static constexpr int P   = Bsz * HW;       // 18432
static constexpr int NH  = 16;
static constexpr int WS  = 8;

#if __has_builtin(__builtin_amdgcn_global_load_async_to_lds_b128)
#define HAVE_ASYNC_LDS 1
#else
#define HAVE_ASYNC_LDS 0
#endif

// ---------------------------------------------------------------------------
// Elementwise converts
// ---------------------------------------------------------------------------
__global__ __launch_bounds__(256) void k_f32_to_bf16(const float* __restrict__ in,
                                                     __bf16* __restrict__ out, int n) {
  int i = blockIdx.x * 256 + threadIdx.x;
  if (i < n) out[i] = (__bf16)in[i];
}

// NCHW fp32 -> CP bf16  (CP: [c][b*HW + hw])
__global__ __launch_bounds__(256) void k_x_to_cp_bf16(const float* __restrict__ x,
                                                      __bf16* __restrict__ out) {
  int i = blockIdx.x * 256 + threadIdx.x;
  if (i >= Bsz * C * HW) return;
  int b  = i / (C * HW);
  int r  = i - b * (C * HW);
  int c  = r / HW;
  int hw = r - c * HW;
  out[(size_t)c * P + b * HW + hw] = (__bf16)x[i];
}

// ---------------------------------------------------------------------------
// bf16 WMMA GEMM: Out[o][p] = sum_c Wm[o][c] * X[c][p],  K = 256
// Block = 256 threads = 8 waves (4 O-tiles x 2 P-tiles) -> 64 O x 32 P tile.
// Weights async-staged to LDS (contiguous 32KB), X transposed to LDS [p][c].
// MODE 0: plain CP store.  MODE 1: fused BN.  MODE 2: NCHW fp32 store (d_out)
// ---------------------------------------------------------------------------
template <int MODE>
__global__ __launch_bounds__(256) void k_gemm_bf16(
    const __bf16* __restrict__ Wm, const __bf16* __restrict__ X,
    float* __restrict__ out,
    const float* __restrict__ bg, const float* __restrict__ bb,
    const float* __restrict__ bm, const float* __restrict__ bv) {
  __shared__ __align__(32) __bf16 lds_w[64 * 256];   // 32KB  [o_local][c]
  __shared__ __align__(32) __bf16 lds_x[32 * 256];   // 16KB  [p_local][c]
  const int tid = threadIdx.x;
  const int p0  = blockIdx.x * 32;
  const int ob0 = blockIdx.y * 64;

  // ---- stage W tile: rows [ob0, ob0+64) are contiguous (64*512B = 32KB) ----
  {
    const __bf16* wsrc = Wm + (size_t)ob0 * 256;
#if HAVE_ASYNC_LDS
#pragma unroll
    for (int i = 0; i < 8; ++i) {
      int e = (tid + i * 256) * 8;   // 8 bf16 = 16B chunks
      __builtin_amdgcn_global_load_async_to_lds_b128(
          (as1_v4i)(wsrc + e), (as3_v4i)(lds_w + e), 0, 0);
    }
#else
#pragma unroll
    for (int i = 0; i < 8; ++i) {
      int e = (tid + i * 256) * 8;
      *(v8bf*)(lds_w + e) = *(const v8bf*)(wsrc + e);
    }
#endif
  }

  // ---- stage X tile transposed: thread t moves row c=t (32 positions) ----
  {
    const __bf16* xsrc = X + (size_t)tid * P + p0;
    v16bf xa = *(const v16bf*)(xsrc);
    v16bf xb = *(const v16bf*)(xsrc + 16);
#pragma unroll
    for (int p = 0; p < 16; ++p) {
      lds_x[p * 256 + tid]        = xa[p];
      lds_x[(p + 16) * 256 + tid] = xb[p];
    }
  }

#if HAVE_ASYNC_LDS
#if __has_builtin(__builtin_amdgcn_s_wait_asynccnt)
  __builtin_amdgcn_s_wait_asynccnt(0);
#else
  asm volatile("s_wait_asynccnt 0x0" ::: "memory");
#endif
#endif
  __syncthreads();

  const int wave = tid >> 5;
  const int lane = tid & 31;
  const int half = lane >> 4;
  const int mn   = lane & 15;
  const int ow   = (wave >> 1) * 16;   // 0,16,32,48  (O sub-tile)
  const int pw   = (wave & 1) * 16;    // 0,16        (P sub-tile)

  v8f acc = {};
#pragma unroll
  for (int kk = 0; kk < 256; kk += 32) {
    union { v16bf v; v8bf h[2]; } A;
    const __bf16* wp = lds_w + (ow + mn) * 256 + kk + half * 8;
    A.h[0] = *(const v8bf*)(wp);        // K = kk + 8*half + e
    A.h[1] = *(const v8bf*)(wp + 16);   // K = kk + 16 + 8*half + e
    union { v16bf v; v8bf h[2]; } B;
    const __bf16* xp = lds_x + (pw + mn) * 256 + kk + half * 8;
    B.h[0] = *(const v8bf*)(xp);        // K = kk + 8*half + e
    B.h[1] = *(const v8bf*)(xp + 16);   // K = kk + 16 + 8*half + e
    acc = __builtin_amdgcn_wmma_f32_16x16x32_bf16(false, A.v, false, B.v,
                                                  (short)0, acc, false, false);
  }

#pragma unroll
  for (int r = 0; r < 8; ++r) {
    int o = ob0 + ow + half * 8 + r;
    int p = p0 + pw + mn;
    float val = acc[r];
    if (MODE == 1) {
      float s = bg[o] * rsqrtf(bv[o] + 1e-5f);
      val = (val - bm[o]) * s + bb[o];
    }
    if (MODE == 2) {
      int b  = p / HW;
      int hw = p - b * HW;
      out[((size_t)b * C + o) * HW + hw] = val;
    } else {
      out[(size_t)o * P + p] = val;
    }
  }
}

// ---------------------------------------------------------------------------
// 3x3 local attention. One block (256 threads = 1 thread/channel) / position.
// ---------------------------------------------------------------------------
__device__ __forceinline__ float block_reduce(float v, float* red, int tid) {
  red[tid] = v;
  __syncthreads();
  for (int s = 128; s > 0; s >>= 1) {
    if (tid < s) red[tid] += red[tid + s];
    __syncthreads();
  }
  float r = red[0];
  __syncthreads();
  return r;
}

__global__ __launch_bounds__(256) void k_local_attn(
    const float* __restrict__ vq, const float* __restrict__ vk,
    const float* __restrict__ vv, const float* __restrict__ locIn,
    const float* __restrict__ rel_x, const float* __restrict__ rel_y,
    const float* __restrict__ l1_bias, float* __restrict__ out) {
  __shared__ float red[256];
  const int p = blockIdx.x;
  const int b = p / HW, hw = p - b * HW, y = hw / W, x = hw - y * W;
  const int c = threadIdx.x;
  const float q = vq[(size_t)c * P + p];

  float logits[9];
#pragma unroll
  for (int k = 0; k < 9; ++k) {
    int dy = k / 3 - 1, dx = k % 3 - 1;
    int yy = y + dy, xx = x + dx;
    float kv = 0.f;
    if (yy >= 0 && yy < H && xx >= 0 && xx < W)
      kv = vk[(size_t)c * P + b * HW + yy * W + xx];
    logits[k] = block_reduce(q * kv, red, c);
  }
  float qx[3], qy[3];
#pragma unroll
  for (int k = 0; k < 3; ++k) {
    float vx = (c < 128) ? q * rel_x[c * 3 + k] : 0.f;
    qx[k] = block_reduce(vx, red, c);
    float vy = (c >= 128) ? q * rel_y[(c - 128) * 3 + k] : 0.f;
    qy[k] = block_reduce(vy, red, c);
  }
  float lg[9], mx = -1e30f;
#pragma unroll
  for (int k = 0; k < 9; ++k) {
    int dy = k / 3, dx = k % 3;
    lg[k] = logits[k] + qx[dx] + qy[dy];
    mx = fmaxf(mx, lg[k]);
  }
  float se = 0.f;
#pragma unroll
  for (int k = 0; k < 9; ++k) { lg[k] = __expf(lg[k] - mx); se += lg[k]; }
  float inv = 1.f / se;
  float acc = 0.f;
#pragma unroll
  for (int k = 0; k < 9; ++k) {
    int dy = k / 3 - 1, dx = k % 3 - 1;
    int yy = y + dy, xx = x + dx;
    float pv = 0.f;
    if (yy >= 0 && yy < H && xx >= 0 && xx < W)
      pv = vv[(size_t)c * P + b * HW + yy * W + xx];
    acc += lg[k] * inv * pv;
  }
  out[(size_t)c * P + p] = locIn[(size_t)c * P + p] + acc + l1_bias[c];
}

// ---------------------------------------------------------------------------
// 8x8 window attention, 16 heads d=16. One block (64 threads) per (window,head).
// ---------------------------------------------------------------------------
__global__ __launch_bounds__(64) void k_win_attn(
    const float* __restrict__ qkv, const float* __restrict__ rel_table,
    float* __restrict__ out) {
  __shared__ float ksh[64][16];
  __shared__ float vsh[64][16];
  const int win  = blockIdx.x;                  // 0..287
  const int head = blockIdx.y;                  // 0..15
  const int b = win / 144, wr = win % 144, wh = wr / 12, ww = wr % 12;
  const int t = threadIdx.x;                    // token in window
  const int ri = t >> 3, ci = t & 7;
  const int h = wh * WS + ri, w = ww * WS + ci;
  const int p = b * HW + h * W + w;

  float q[16];
#pragma unroll
  for (int d = 0; d < 16; ++d) {
    int ch = head * 16 + d;
    q[d]      = qkv[(size_t)ch * P + p];
    ksh[t][d] = qkv[(size_t)(256 + ch) * P + p];
    vsh[t][d] = qkv[(size_t)(512 + ch) * P + p];
  }
  __syncthreads();

  float dots[64], mx = -1e30f;
  for (int j = 0; j < 64; ++j) {
    float s = 0.f;
#pragma unroll
    for (int d = 0; d < 16; ++d) s += q[d] * ksh[j][d];
    int rj = j >> 3, cj = j & 7;
    int idx = (ri - rj + 7) * 15 + (ci - cj + 7);
    s = s * 0.25f + rel_table[idx * NH + head];
    dots[j] = s;
    mx = fmaxf(mx, s);
  }
  float se = 0.f;
  for (int j = 0; j < 64; ++j) { dots[j] = __expf(dots[j] - mx); se += dots[j]; }
  float inv = 1.f / se;
  float o[16] = {};
  for (int j = 0; j < 64; ++j) {
    float a = dots[j] * inv;
#pragma unroll
    for (int d = 0; d < 16; ++d) o[d] += a * vsh[j][d];
  }
#pragma unroll
  for (int d = 0; d < 16; ++d) out[(size_t)(head * 16 + d) * P + p] = o[d];
}

// ---------------------------------------------------------------------------
// Directional max/avg pooling (window 8, reflect(0,1), pad 3/3) + add local.
// ---------------------------------------------------------------------------
__global__ __launch_bounds__(256) void k_pool_add(
    const float* __restrict__ attn, const float* __restrict__ loc,
    float* __restrict__ out) {
  int idx = blockIdx.x * 256 + threadIdx.x;
  if (idx >= C * P) return;
  int c = idx / P, p = idx - c * P;
  int b = p / HW, hw = p - b * HW, y = hw / W, x = hw - y * W;
  const float* base = attn + (size_t)c * P + b * HW;

  float mh = -1e30f, sh = 0.f, mw = -1e30f, sw = 0.f;
#pragma unroll
  for (int k = 0; k < 8; ++k) {
    int j = y - 3 + k;
    if (j >= 0 && j <= H) {
      int jj = (j == H) ? H - 2 : j;
      float v = base[jj * W + x];
      mh = fmaxf(mh, v);
      sh += v;
    }
    int i = x - 3 + k;
    if (i >= 0 && i <= W) {
      int ii = (i == W) ? W - 2 : i;
      float v = base[y * W + ii];
      mw = fmaxf(mw, v);
      sw += v;
    }
  }
  out[idx] = mh + mw + (sh + sw) * 0.125f + loc[idx];
}

// ---------------------------------------------------------------------------
// 8x8 depthwise conv (reflect(0,1) then zero pad 3/3) + BN(bnp) -> bf16 CP
// ---------------------------------------------------------------------------
__global__ __launch_bounds__(256) void k_dwconv_bn(
    const float* __restrict__ pre, const float* __restrict__ dww,
    const float* __restrict__ bg, const float* __restrict__ bb,
    const float* __restrict__ bm, const float* __restrict__ bv,
    __bf16* __restrict__ out) {
  int idx = blockIdx.x * 256 + threadIdx.x;
  if (idx >= C * P) return;
  int c = idx / P, p = idx - c * P;
  int b = p / HW, hw = p - b * HW, y = hw / W, x = hw - y * W;
  const float* base = pre + (size_t)c * P + b * HW;
  const float* wk = dww + c * 64;

  float acc = 0.f;
#pragma unroll
  for (int ky = 0; ky < 8; ++ky) {
    int iy = y - 3 + ky;
    if (iy < 0 || iy > H) continue;
    int iyy = (iy == H) ? H - 2 : iy;
#pragma unroll
    for (int kx = 0; kx < 8; ++kx) {
      int ix = x - 3 + kx;
      if (ix < 0 || ix > W) continue;
      int ixx = (ix == W) ? W - 2 : ix;
      acc += base[iyy * W + ixx] * wk[ky * 8 + kx];
    }
  }
  float s = bg[c] * rsqrtf(bv[c] + 1e-5f);
  out[idx] = (__bf16)((acc - bm[c]) * s + bb[c]);
}

// ---------------------------------------------------------------------------
extern "C" void kernel_launch(void* const* d_in, const int* in_sizes, int n_in,
                              void* d_out, int out_size, void* d_ws, size_t ws_size,
                              hipStream_t stream) {
  const float* x      = (const float*)d_in[0];
  const float* qkv_w  = (const float*)d_in[1];
  const float* l2_w   = (const float*)d_in[2];
  const float* bn2_g  = (const float*)d_in[3];
  const float* bn2_b  = (const float*)d_in[4];
  const float* bn2_m  = (const float*)d_in[5];
  const float* bn2_v  = (const float*)d_in[6];
  const float* wq     = (const float*)d_in[7];
  const float* wk     = (const float*)d_in[8];
  const float* wv     = (const float*)d_in[9];
  const float* rel_x  = (const float*)d_in[10];
  const float* rel_y  = (const float*)d_in[11];
  const float* l1_b   = (const float*)d_in[12];
  const float* rel_t  = (const float*)d_in[13];
  const float* dw_w   = (const float*)d_in[14];
  const float* bnp_g  = (const float*)d_in[15];
  const float* bnp_b  = (const float*)d_in[16];
  const float* bnp_m  = (const float*)d_in[17];
  const float* bnp_v  = (const float*)d_in[18];
  const float* pw_w   = (const float*)d_in[19];

  // workspace carve (256B aligned)
  size_t off = 0;
  auto carve = [&](size_t bytes) -> void* {
    void* r = (char*)d_ws + off;
    off += (bytes + 255) & ~(size_t)255;
    return r;
  };
  __bf16* xbf   = (__bf16*)carve((size_t)C * P * 2);
  __bf16* w_qkv = (__bf16*)carve((size_t)3 * C * C * 2);
  __bf16* w_l2  = (__bf16*)carve((size_t)C * C * 2);
  __bf16* w_q   = (__bf16*)carve((size_t)C * C * 2);
  __bf16* w_k   = (__bf16*)carve((size_t)C * C * 2);
  __bf16* w_v   = (__bf16*)carve((size_t)C * C * 2);
  __bf16* w_pw  = (__bf16*)carve((size_t)C * C * 2);
  float* localb = (float*)carve((size_t)C * P * 4);
  float* local2 = (float*)carve((size_t)C * P * 4);
  float* vqb    = (float*)carve((size_t)C * P * 4);
  float* vkb    = (float*)carve((size_t)C * P * 4);
  float* vvb    = (float*)carve((size_t)C * P * 4);
  float* qkvb   = (float*)carve((size_t)3 * C * P * 4);
  float* attnb  = (float*)carve((size_t)C * P * 4);
  float* preb   = (float*)carve((size_t)C * P * 4);
  __bf16* dwbf  = (__bf16*)carve((size_t)C * P * 2);
  (void)ws_size;

  // --- conversions ---
  k_x_to_cp_bf16<<<(Bsz * C * HW + 255) / 256, 256, 0, stream>>>(x, xbf);
  k_f32_to_bf16<<<(3 * C * C + 255) / 256, 256, 0, stream>>>(qkv_w, w_qkv, 3 * C * C);
  k_f32_to_bf16<<<(C * C + 255) / 256, 256, 0, stream>>>(l2_w, w_l2, C * C);
  k_f32_to_bf16<<<(C * C + 255) / 256, 256, 0, stream>>>(wq, w_q, C * C);
  k_f32_to_bf16<<<(C * C + 255) / 256, 256, 0, stream>>>(wk, w_k, C * C);
  k_f32_to_bf16<<<(C * C + 255) / 256, 256, 0, stream>>>(wv, w_v, C * C);
  k_f32_to_bf16<<<(C * C + 255) / 256, 256, 0, stream>>>(pw_w, w_pw, C * C);

  // --- WMMA GEMMs ---
  dim3 blk(256);
  dim3 g256(P / 32, C / 64);         // (576, 4)
  dim3 g768(P / 32, 3 * C / 64);     // (576, 12)
  k_gemm_bf16<1><<<g256, blk, 0, stream>>>(w_l2, xbf, localb, bn2_g, bn2_b, bn2_m, bn2_v);
  k_gemm_bf16<0><<<g256, blk, 0, stream>>>(w_q, xbf, vqb, nullptr, nullptr, nullptr, nullptr);
  k_gemm_bf16<0><<<g256, blk, 0, stream>>>(w_k, xbf, vkb, nullptr, nullptr, nullptr, nullptr);
  k_gemm_bf16<0><<<g256, blk, 0, stream>>>(w_v, xbf, vvb, nullptr, nullptr, nullptr, nullptr);
  k_gemm_bf16<0><<<g768, blk, 0, stream>>>(w_qkv, xbf, qkvb, nullptr, nullptr, nullptr, nullptr);

  // --- local 3x3 attention ---
  k_local_attn<<<P, 256, 0, stream>>>(vqb, vkb, vvb, localb, rel_x, rel_y, l1_b, local2);

  // --- window attention ---
  k_win_attn<<<dim3(Bsz * 12 * 12, NH), 64, 0, stream>>>(qkvb, rel_t, attnb);

  // --- pooling + add local ---
  k_pool_add<<<(C * P + 255) / 256, 256, 0, stream>>>(attnb, local2, preb);

  // --- depthwise 8x8 + BN -> bf16 ---
  k_dwconv_bn<<<(C * P + 255) / 256, 256, 0, stream>>>(preb, dw_w, bnp_g, bnp_b,
                                                       bnp_m, bnp_v, dwbf);

  // --- final pointwise GEMM, NCHW fp32 store ---
  k_gemm_bf16<2><<<g256, blk, 0, stream>>>(w_pw, dwbf, (float*)d_out,
                                           nullptr, nullptr, nullptr, nullptr);
}
